// CrossAttentionBlock_66228395704446
// MI455X (gfx1250) — matrix operand order repaired and checked by reading, compiled, not run on hardware
//
#include <hip/hip_runtime.h>
#include <hip/hip_bf16.h>

typedef __attribute__((ext_vector_type(16))) _Float16 v16h;
typedef __attribute__((ext_vector_type(8)))  _Float16 v8h;
typedef __attribute__((ext_vector_type(8)))  float    v8f;
typedef __attribute__((ext_vector_type(4)))  unsigned int u32x4;
typedef __attribute__((ext_vector_type(8)))  int i32x8;
typedef __attribute__((ext_vector_type(4)))  int i32x4;

#define WMMA_F16(A, B, Cacc) \
  __builtin_amdgcn_wmma_f32_16x16x32_f16(false, (A), false, (B), (short)0, (Cacc), false, false)

#if defined(__has_builtin)
#if __has_builtin(__builtin_amdgcn_tensor_load_to_lds) && __has_builtin(__builtin_amdgcn_s_wait_tensorcnt)
#define HAVE_TDM 1
#endif
#endif
#ifndef HAVE_TDM
#define HAVE_TDM 0
#endif

#if HAVE_TDM
// Issue one TDM 2-D tile load: global (row-major, f16 elements) -> LDS (linear).
// Dims/strides in elements (data_size units). D# layout per CDNA5 ISA ch.8.
__device__ __forceinline__ void tdm_load_2d(unsigned lds_addr, const void* gptr,
                                            unsigned tile_d0, unsigned tile_d1,
                                            unsigned tensor_d0, unsigned tensor_d1,
                                            unsigned long long stride0) {
  unsigned long long ga = (unsigned long long)gptr;
  u32x4 g0;
  g0[0] = 1u;                                            // count=1, user descriptor
  g0[1] = lds_addr;                                      // LDS byte address
  g0[2] = (unsigned)(ga & 0xFFFFFFFFu);                  // global_addr[31:0]
  g0[3] = (unsigned)((ga >> 32) & 0x01FFFFFFu) | (2u << 30);  // global_addr[56:32] | type=2
  i32x8 g1;
  g1[0] = (int)(1u << 16);                               // workgroup_mask=0, data_size=1 (2B)
  g1[1] = (int)((tensor_d0 & 0xFFFFu) << 16);            // abar_addr=0 | tensor_dim0[15:0]
  g1[2] = (int)((tensor_d0 >> 16) | ((tensor_d1 & 0xFFFFu) << 16));
  g1[3] = (int)((tensor_d1 >> 16) | ((tile_d0 & 0xFFFFu) << 16));
  g1[4] = (int)(tile_d1 & 0xFFFFu);                      // tile_dim1 | tile_dim2=0
  g1[5] = (int)(unsigned)(stride0 & 0xFFFFFFFFu);        // tensor_dim0_stride[31:0]
  g1[6] = (int)(unsigned)((stride0 >> 32) & 0xFFFFu);    // stride0[47:32] | stride1_lo=0
  g1[7] = 0;
  i32x4 z4 = {};
#if __clang_major__ >= 23
  i32x8 z8 = {};
  __builtin_amdgcn_tensor_load_to_lds(g0, g1, z4, z4, z8, 0);
#else
  __builtin_amdgcn_tensor_load_to_lds(g0, g1, z4, z4, 0);
#endif
}
#endif

// ---------------------------------------------------------------------------
// Kernel 1: GroupNorm statistics. One block per (src, b, g). src 0 = x, 1 = ctx.
// stats layout: [src][b][g][2] = {mu, rstd}
// ---------------------------------------------------------------------------
__global__ void gn_stats_kernel(const float* __restrict__ x,
                                const float* __restrict__ ctx,
                                float* __restrict__ stats) {
  const int HW = 4096, C = 256, CPG = 8;
  int id  = blockIdx.x;        // 0..255  (2 * B * G)
  int src = id >> 7;
  int bg  = id & 127;
  int b   = bg >> 5;
  int g   = bg & 31;
  const float* p = (src == 0) ? x : ctx;
  size_t base = ((size_t)b * C + (size_t)g * CPG) * HW;   // 8 channels contiguous

  float s = 0.f, ss = 0.f;
  for (int i = threadIdx.x; i < CPG * HW; i += blockDim.x) {
    float v = p[base + i];
    s += v; ss += v * v;
  }
  #pragma unroll
  for (int off = 16; off > 0; off >>= 1) {
    s  += __shfl_xor(s,  off, 32);
    ss += __shfl_xor(ss, off, 32);
  }
  __shared__ float red[2][8];
  int wave = threadIdx.x >> 5;
  if ((threadIdx.x & 31) == 0) { red[0][wave] = s; red[1][wave] = ss; }
  __syncthreads();
  if (threadIdx.x == 0) {
    float S = 0.f, SS = 0.f;
    for (int w = 0; w < 8; ++w) { S += red[0][w]; SS += red[1][w]; }
    float inv_n = 1.0f / (float)(CPG * HW);
    float mu  = S * inv_n;
    float var = SS * inv_n - mu * mu;
    stats[(id << 1) + 0] = mu;
    stats[(id << 1) + 1] = rsqrtf(var + 1e-5f);
  }
}

// ---------------------------------------------------------------------------
// Kernel 2: apply GN, convert to f16, re-layout token-major [B][HW][C].
// ---------------------------------------------------------------------------
__global__ void pack_kernel(const float* __restrict__ x,
                            const float* __restrict__ ctx,
                            const float* __restrict__ gqw, const float* __restrict__ gqb,
                            const float* __restrict__ gcw, const float* __restrict__ gcb,
                            const float* __restrict__ stats,
                            _Float16* __restrict__ xn_h,
                            _Float16* __restrict__ cn_h,
                            _Float16* __restrict__ ctx_h) {
  const int C = 256, HW = 4096;
  const size_t total = (size_t)4 * C * HW;     // 2^22
  for (size_t idx = (size_t)blockIdx.x * blockDim.x + threadIdx.x; idx < total;
       idx += (size_t)gridDim.x * blockDim.x) {
    int s = (int)(idx & (HW - 1));
    int c = (int)((idx >> 12) & (C - 1));
    int b = (int)(idx >> 20);
    int g = c >> 3;
    size_t dst = ((size_t)b * HW + s) * C + c;  // token-major

    const float* stx = stats + ((((0 * 4 + b) * 32) + g) << 1);
    const float* stc = stats + ((((1 * 4 + b) * 32) + g) << 1);
    float xv = x[idx];
    float cv = ctx[idx];
    xn_h[dst]  = (_Float16)((xv - stx[0]) * stx[1] * gqw[c] + gqb[c]);
    cn_h[dst]  = (_Float16)((cv - stc[0]) * stc[1] * gcw[c] + gcb[c]);
    ctx_h[dst] = (_Float16)cv;
  }
}

// ---------------------------------------------------------------------------
// Kernel 3: weights -> f16, concatenated [Wq|Wk|Wv|Wo], each [256][256] row-major.
// ---------------------------------------------------------------------------
__global__ void wcvt_kernel(const float* __restrict__ wq, const float* __restrict__ wk,
                            const float* __restrict__ wv, const float* __restrict__ wo,
                            _Float16* __restrict__ wh) {
  const int n = 256 * 256;
  for (int i = blockIdx.x * blockDim.x + threadIdx.x; i < 4 * n;
       i += gridDim.x * blockDim.x) {
    int m = i / n, r = i - m * n;
    const float* src = (m == 0) ? wq : (m == 1) ? wk : (m == 2) ? wv : wo;
    wh[i] = (_Float16)src[r];
  }
}

// ---------------------------------------------------------------------------
// Kernel 4: QKV projection GEMM. One 16x16 D tile per wave.
// ---------------------------------------------------------------------------
__global__ __launch_bounds__(128)
void proj_kernel(const _Float16* __restrict__ wh,
                 const _Float16* __restrict__ xn_h,
                 const _Float16* __restrict__ cn_h,
                 const _Float16* __restrict__ ctx_h,
                 _Float16* __restrict__ q_h,
                 _Float16* __restrict__ k_h,
                 _Float16* __restrict__ v_h) {
  const int C = 256, HW = 4096;
  int lane = threadIdx.x & 31;
  int wave = threadIdx.x >> 5;
  int mat  = blockIdx.z;                 // 0=q, 1=k, 2=v
  int b    = blockIdx.y;
  int tile = blockIdx.x * 4 + wave;      // 0..4095
  int mtile = tile & 15;
  int ntile = tile >> 4;

  const _Float16* W  = wh + (size_t)mat * C * C;
  const _Float16* In = (mat == 0) ? xn_h : (mat == 1) ? cn_h : ctx_h;
  In += (size_t)b * HW * C;

  int m  = lane & 15;
  int hi = lane >> 4;

  v8f acc = {};
  #pragma unroll
  for (int kc = 0; kc < 8; ++kc) {
    const _Float16* arow = W + (size_t)(mtile * 16 + m) * C + kc * 32;
    union { v16h v; v8h h[2]; } ua;
    ua.h[0] = *(const v8h*)(arow + hi * 16);
    ua.h[1] = *(const v8h*)(arow + hi * 16 + 8);
    const _Float16* brow = In + (size_t)(ntile * 16 + m) * C + kc * 32 + hi * 16;
    v16h bv = *(const v16h*)brow;
    acc = WMMA_F16(ua.v, bv, acc);
  }
  #pragma unroll
  for (int r = 0; r < 8; ++r) {
    int o = mtile * 16 + r + 8 * hi;
    int s = ntile * 16 + m;
    int head = o >> 6, ch = o & 63;
    _Float16 val = (_Float16)acc[r];
    if (mat == 2)
      v_h[(((size_t)b * 4 + head) * 64 + ch) * HW + s] = val;
    else if (mat == 0)
      q_h[(((size_t)b * 4 + head) * HW + s) * 64 + ch] = val;
    else
      k_h[(((size_t)b * 4 + head) * HW + s) * 64 + ch] = val;
  }
}

// ---------------------------------------------------------------------------
// Kernel 5: flash attention. Block = 4 waves; wave owns 16 queries.
// K/V chunks (32 keys) staged in double-buffered LDS via the Tensor Data Mover
// (TENSOR_LOAD_TO_LDS, tracked with TENSORcnt); P staged in per-wave LDS.
// ---------------------------------------------------------------------------
__global__ __launch_bounds__(128)
void attn_kernel(const _Float16* __restrict__ q_h,
                 const _Float16* __restrict__ k_h,
                 const _Float16* __restrict__ v_h,
                 _Float16* __restrict__ ao_h) {   // [B][HW][C] f16
  const int HW = 4096, HD = 64, C = 256;
  int lane = threadIdx.x & 31;
  int wave = threadIdx.x >> 5;
  int bh   = blockIdx.y;                 // b*4 + head
  int b    = bh >> 2;
  int head = bh & 3;
  int q0   = blockIdx.x * 64 + wave * 16;

  const _Float16* Q = q_h + (size_t)bh * HW * HD;
  const _Float16* K = k_h + (size_t)bh * HW * HD;
  const _Float16* V = v_h + (size_t)bh * HD * HW;

  int m  = lane & 15;
  int hi = lane >> 4;

  // Q A-frags, held in VGPRs for the whole key loop
  v16h aq[2];
  #pragma unroll
  for (int kc = 0; kc < 2; ++kc) {
    const _Float16* r = Q + (size_t)(q0 + m) * HD + kc * 32;
    union { v16h v; v8h h[2]; } u;
    u.h[0] = *(const v8h*)(r + hi * 16);
    u.h[1] = *(const v8h*)(r + hi * 16 + 8);
    aq[kc] = u.v;
  }

  float mrow[8], lrow[8];
  v8f oacc[4];
  #pragma unroll
  for (int r = 0; r < 8; ++r) { mrow[r] = -3.0e38f; lrow[r] = 0.f; }
  #pragma unroll
  for (int c = 0; c < 4; ++c) { v8f z = {}; oacc[c] = z; }

  // Double-buffered K (32 keys x 64 ch) and V (64 ch x 32 keys) tiles + P tiles.
  __shared__ __attribute__((aligned(64))) _Float16 lds_k[2][32][64];
  __shared__ __attribute__((aligned(64))) _Float16 lds_v[2][64][32];
  __shared__ __attribute__((aligned(32))) _Float16 lds_p[4][16][32];

  const float scale = 0.125f;  // 1/sqrt(64)

#if HAVE_TDM
  if (wave == 0) {
    tdm_load_2d((unsigned)(unsigned long long)(const void*)&lds_k[0][0][0],
                K, /*tile_d0=*/64, /*tile_d1=*/32, /*tensor_d0=*/64, /*tensor_d1=*/HW,
                /*stride0=*/64);
    tdm_load_2d((unsigned)(unsigned long long)(const void*)&lds_v[0][0][0],
                V, /*tile_d0=*/32, /*tile_d1=*/64, /*tensor_d0=*/HW, /*tensor_d1=*/64,
                /*stride0=*/HW);
  }
#endif

  for (int kk = 0; kk < HW; kk += 32) {
    int cur = (kk >> 5) & 1;
#if HAVE_TDM
    __builtin_amdgcn_s_wait_tensorcnt(0);
    __syncthreads();                       // tile `cur` visible to all waves
    if (wave == 0 && (kk + 32) < HW) {     // DMA next chunk into the other buffer
      tdm_load_2d((unsigned)(unsigned long long)(const void*)&lds_k[cur ^ 1][0][0],
                  K + (size_t)(kk + 32) * HD, 64, 32, 64, HW, 64);
      tdm_load_2d((unsigned)(unsigned long long)(const void*)&lds_v[cur ^ 1][0][0],
                  V + (size_t)(kk + 32), 32, 64, HW, 64, HW);
    }
#else
    // Cooperative fallback: 128 threads stage K/V chunk with b128 loads/stores.
    for (int i = threadIdx.x; i < 32 * 64 / 8; i += blockDim.x) {
      int row = i >> 3, cv8 = i & 7;
      *(v8h*)&lds_k[cur][row][cv8 * 8] =
          *(const v8h*)(K + (size_t)(kk + row) * HD + cv8 * 8);
    }
    for (int i = threadIdx.x; i < 64 * 32 / 8; i += blockDim.x) {
      int row = i >> 2, cv8 = i & 3;
      *(v8h*)&lds_v[cur][row][cv8 * 8] =
          *(const v8h*)(V + (size_t)row * HW + kk + cv8 * 8);
    }
    __syncthreads();
#endif

    // S tiles: t=0 -> keys kk..kk+15, t=1 -> keys kk+16..kk+31 (B-frags from LDS)
    v8f s[2];
    #pragma unroll
    for (int t = 0; t < 2; ++t) {
      v8f accv = {};
      #pragma unroll
      for (int kc = 0; kc < 2; ++kc) {
        v16h bv = *(const v16h*)&lds_k[cur][t * 16 + m][kc * 32 + hi * 16];
        accv = WMMA_F16(aq[kc], bv, accv);
      }
      s[t] = accv;
    }
    // Online softmax. Row M = r + 8*hi spans lanes (hi*16 .. hi*16+15) in N.
    #pragma unroll
    for (int r = 0; r < 8; ++r) {
      float s0 = s[0][r] * scale;
      float s1 = s[1][r] * scale;
      float rmax = fmaxf(s0, s1);
      #pragma unroll
      for (int off = 8; off > 0; off >>= 1)
        rmax = fmaxf(rmax, __shfl_xor(rmax, off, 32));
      float newm  = fmaxf(mrow[r], rmax);
      float alpha = __expf(mrow[r] - newm);
      float p0 = __expf(s0 - newm);
      float p1 = __expf(s1 - newm);
      float rsum = p0 + p1;
      #pragma unroll
      for (int off = 8; off > 0; off >>= 1)
        rsum += __shfl_xor(rsum, off, 32);
      lrow[r] = lrow[r] * alpha + rsum;
      mrow[r] = newm;
      #pragma unroll
      for (int c = 0; c < 4; ++c) oacc[c][r] *= alpha;
      int row = r + 8 * hi;
      lds_p[wave][row][m]      = (_Float16)p0;
      lds_p[wave][row][m + 16] = (_Float16)p1;
    }
    // P as A-frag [16 q x 32 keys] from this wave's LDS region
    union { v16h v; v8h h[2]; } up;
    up.h[0] = *(const v8h*)&lds_p[wave][m][hi * 16];
    up.h[1] = *(const v8h*)&lds_p[wave][m][hi * 16 + 8];
    // PV: 4 head-dim chunks of 16 channels (V B-frags from LDS)
    #pragma unroll
    for (int c = 0; c < 4; ++c) {
      v16h bv = *(const v16h*)&lds_v[cur][c * 16 + m][hi * 16];
      oacc[c] = WMMA_F16(up.v, bv, oacc[c]);
    }
    __syncthreads();   // everyone done reading buffer `cur` before it is refilled
  }

  // Normalize by l and store token-major f16 for the Wo GEMM
  #pragma unroll
  for (int r = 0; r < 8; ++r) {
    float inv = 1.0f / lrow[r];
    int qi = q0 + r + 8 * hi;
    #pragma unroll
    for (int c = 0; c < 4; ++c) {
      int ch = head * 64 + c * 16 + m;
      ao_h[((size_t)b * HW + qi) * C + ch] = (_Float16)(oacc[c][r] * inv);
    }
  }
}

// ---------------------------------------------------------------------------
// Kernel 6: output projection + bias + residual -> fp32 out [B][C][HW].
// ---------------------------------------------------------------------------
__global__ __launch_bounds__(128)
void outproj_kernel(const _Float16* __restrict__ wo_h,
                    const _Float16* __restrict__ ao_h,   // [B][HW][C]
                    const float* __restrict__ x,
                    const float* __restrict__ bo,
                    float* __restrict__ out) {
  const int C = 256, HW = 4096;
  int lane = threadIdx.x & 31;
  int wave = threadIdx.x >> 5;
  int b    = blockIdx.y;
  int tile = blockIdx.x * 4 + wave;
  int mtile = tile & 15;
  int ntile = tile >> 4;
  const _Float16* In = ao_h + (size_t)b * HW * C;
  int m  = lane & 15;
  int hi = lane >> 4;

  v8f acc = {};
  #pragma unroll
  for (int kc = 0; kc < 8; ++kc) {
    const _Float16* arow = wo_h + (size_t)(mtile * 16 + m) * C + kc * 32;
    union { v16h v; v8h h[2]; } ua;
    ua.h[0] = *(const v8h*)(arow + hi * 16);
    ua.h[1] = *(const v8h*)(arow + hi * 16 + 8);
    const _Float16* brow = In + (size_t)(ntile * 16 + m) * C + kc * 32 + hi * 16;
    v16h bv = *(const v16h*)brow;
    acc = WMMA_F16(ua.v, bv, acc);
  }
  #pragma unroll
  for (int r = 0; r < 8; ++r) {
    int o = mtile * 16 + r + 8 * hi;
    int s = ntile * 16 + m;
    size_t idx = ((size_t)b * C + o) * HW + s;
    out[idx] = x[idx] + acc[r] + bo[o];
  }
}

// ---------------------------------------------------------------------------
// Host launcher
// ---------------------------------------------------------------------------
extern "C" void kernel_launch(void* const* d_in, const int* in_sizes, int n_in,
                              void* d_out, int out_size, void* d_ws, size_t ws_size,
                              hipStream_t stream) {
  (void)in_sizes; (void)n_in; (void)out_size; (void)ws_size;
  const float* x   = (const float*)d_in[0];
  const float* ctx = (const float*)d_in[1];
  const float* gqw = (const float*)d_in[2];
  const float* gqb = (const float*)d_in[3];
  const float* gcw = (const float*)d_in[4];
  const float* gcb = (const float*)d_in[5];
  const float* Wq  = (const float*)d_in[6];
  const float* Wk  = (const float*)d_in[7];
  const float* Wv  = (const float*)d_in[8];
  const float* Wo  = (const float*)d_in[9];
  const float* bo  = (const float*)d_in[10];
  float* out = (float*)d_out;

  char* ws = (char*)d_ws;
  size_t off = 0;
  auto wsalloc = [&](size_t bytes) -> void* {
    void* p = ws + off;
    off = (off + bytes + 255) & ~(size_t)255;
    return p;
  };
  const size_t TOK = (size_t)4 * 4096 * 256;   // B*HW*C elements
  float*    stats = (float*)wsalloc(512 * sizeof(float));
  _Float16* wh    = (_Float16*)wsalloc((size_t)4 * 65536 * sizeof(_Float16));
  _Float16* xn_h  = (_Float16*)wsalloc(TOK * 2);
  _Float16* cn_h  = (_Float16*)wsalloc(TOK * 2);
  _Float16* ctx_h = (_Float16*)wsalloc(TOK * 2);
  _Float16* q_h   = (_Float16*)wsalloc(TOK * 2);
  _Float16* k_h   = (_Float16*)wsalloc(TOK * 2);
  _Float16* v_h   = (_Float16*)wsalloc(TOK * 2);
  _Float16* ao_h  = (_Float16*)wsalloc(TOK * 2);

  gn_stats_kernel<<<256, 256, 0, stream>>>(x, ctx, stats);
  pack_kernel<<<2048, 256, 0, stream>>>(x, ctx, gqw, gqb, gcw, gcb, stats,
                                        xn_h, cn_h, ctx_h);
  wcvt_kernel<<<512, 256, 0, stream>>>(Wq, Wk, Wv, Wo, wh);
  proj_kernel<<<dim3(1024, 4, 3), 128, 0, stream>>>(wh, xn_h, cn_h, ctx_h,
                                                    q_h, k_h, v_h);
  attn_kernel<<<dim3(64, 16), 128, 0, stream>>>(q_h, k_h, v_h, ao_h);
  outproj_kernel<<<dim3(1024, 4), 128, 0, stream>>>(wh + (size_t)3 * 65536,
                                                    ao_h, x, bo, out);
}
